// proximity_conv_76845554860269
// MI455X (gfx1250) — matrix operand order repaired and checked by reading, not compiled, not run on hardware
//
#include <hip/hip_runtime.h>

typedef __attribute__((ext_vector_type(16))) _Float16 v16h;
typedef __attribute__((ext_vector_type(8)))  float    v8f;

#define B_      8
#define C_      64
#define O_      64
#define H_      96
#define W_      96
#define HW_     (H_ * W_)
#define KS_     576          // K dimension = C*9
#define KSTEPS  18           // 576 / 32
#define PIXWG   128          // pixels per workgroup (8 waves x 16)
#define THREADS 256

// ---------------------------------------------------------------------------
// Prep: weight [64, 576] f32 -> f16, pre-swizzled into CDNA5 WMMA A-fragment
// layout (16x32 f16, ISA 7.12.2):  k = 2*(v%4)+h + (v>=4?16:0) + (lane>=16?8:0)
// Stored as [frag = kstep*4+mtile][lane(32)][e(16)] halves (32B per lane/frag).
// ---------------------------------------------------------------------------
__global__ void prep_weight_frag(const float* __restrict__ w,
                                 _Float16* __restrict__ wsA) {
  int gid = blockIdx.x * blockDim.x + threadIdx.x;
  if (gid >= KSTEPS * 4 * 32 * 16) return;
  int e     = gid & 15;
  int lane  = (gid >> 4) & 31;
  int frag  = gid >> 9;
  int mtile = frag & 3;
  int kstep = frag >> 2;
  int v  = e >> 1;
  int hh = e & 1;
  int kk = 2 * (v & 3) + hh + ((v >= 4) ? 16 : 0) + ((lane >= 16) ? 8 : 0);
  int m  = mtile * 16 + (lane & 15);
  int k  = kstep * 32 + kk;
  wsA[gid] = (_Float16)w[m * KS_ + k];
}

// ---------------------------------------------------------------------------
// Fused: KNN selection (top-9 of 25 by |intensity diff|) + gather-GEMM via WMMA
// ---------------------------------------------------------------------------
__global__ __launch_bounds__(THREADS) void pconv_knn_wmma(
    const float* __restrict__ x,        // [B, 64, 96, 96]
    const _Float16* __restrict__ wsA,   // pre-swizzled weight fragments (72KB)
    const float* __restrict__ meanp,
    const float* __restrict__ stdp,
    float* __restrict__ out) {          // [B, 64, 96, 96]
  __shared__ __align__(32) _Float16 wA[KSTEPS * 4 * 32 * 16];  // 72 KB
  __shared__ int nbrOff[PIXWG * 9];                            // 4.5 KB

  const int tid   = threadIdx.x;
  const int base  = blockIdx.x * PIXWG;       // global pixel id over B*H*W
  const int b     = base / HW_;
  const int pbase = base - b * HW_;           // within-image base (mult of 128)
  const float* xb = x + (size_t)b * C_ * HW_;

  // ---- Phase 1: stage pre-swizzled weight fragments into LDS (uint4 copy)
  {
    const uint4* src = (const uint4*)wsA;
    uint4* dst = (uint4*)wA;
#pragma unroll
    for (int i = tid; i < (KSTEPS * 4 * 32 * 16 * 2) / 16; i += THREADS)
      dst[i] = src[i];
  }

  // ---- Phase 2: per-pixel neighbor selection (one lane per pixel, 128 px)
  if (tid < PIXWG) {
    const float sd = stdp[0], mn = meanp[0];
    const int p  = pbase + tid;
    const int ph = p / W_;
    const int pw = p - ph * W_;
    const float c0 = xb[p] * sd + mn;     // center intensity (denormalized)
    float dif[25];
    int   offs[25];
#pragma unroll
    for (int t = 0; t < 25; ++t) {
      const int dy = t / 5 - 2, dx = t % 5 - 2;
      const int hy = ph + dy, wx = pw + dx;
      const bool val = (hy >= 0) & (hy < H_) & (wx >= 0) & (wx < W_);
      const int o = val ? (hy * W_ + wx) : -1;
      const float v = val ? (xb[o] * sd + mn) : 0.0f;   // zero-pad candidate
      dif[t]  = (t == 12) ? -1.0f : fabsf(v - c0);      // center forced first
      offs[t] = o;
    }
    // 9-way selection; strict '<' -> lower index wins ties (matches top_k)
    unsigned taken = 0;
    for (int s = 0; s < 9; ++s) {
      float best = 3.0e38f;
      int bi = 0, boff = -1;
#pragma unroll
      for (int t = 0; t < 25; ++t) {
        const bool c = (((taken >> t) & 1u) == 0u) && (dif[t] < best);
        best = c ? dif[t]  : best;
        bi   = c ? t       : bi;
        boff = c ? offs[t] : boff;
      }
      taken |= (1u << bi);
      nbrOff[tid * 9 + s] = boff;         // -1 == zero-padded neighbor
    }
  }
  __syncthreads();

  // ---- Phase 3: each wave -> 64x16 output tile, K = 576 in 18 steps of 32
  const int lane   = tid & 31;
  const int waveId = tid >> 5;
  const int nIdx   = lane & 15;                   // pixel column within tile
  const int kb     = (lane < 16) ? 0 : 8;         // B-fragment K-half per lane
  const int pix9   = (waveId * 16 + nIdx) * 9;

  v8f acc[4] = {};                                // 4 M-tiles of 16 channels

  for (int kstep = 0; kstep < KSTEPS; ++kstep) {
    // Build B fragment (32x16 f16): lane holds K kb..kb+7 and 16+kb..16+kb+7 for its pixel
    v16h bf;
    const int k0 = kstep * 32 + kb;
#pragma unroll
    for (int e = 0; e < 16; ++e) {
      const unsigned k = (unsigned)(k0 + e + ((e >= 8) ? 8 : 0));
      const unsigned c = k / 9u;                  // channel
      const unsigned j = k - c * 9u;              // top-k rank
      const int o  = nbrOff[pix9 + (int)j];       // LDS gather of offset
      const int so = (o >= 0) ? o : 0;            // safe address for pads
      float v = xb[(size_t)c * HW_ + so];         // L2-resident dword gather
      v = (o >= 0) ? v : 0.0f;                    // mask padded neighbors
      bf[e] = (_Float16)v;
    }
    // 4 WMMAs: A fragments streamed from LDS (pre-swizzled layout)
#pragma unroll
    for (int mt = 0; mt < 4; ++mt) {
      const v16h af = *(const v16h*)&wA[((kstep * 4 + mt) * 32 + lane) * 16];
      acc[mt] = __builtin_amdgcn_wmma_f32_16x16x32_f16(
          /*neg_a=*/false, af, /*neg_b=*/false, bf,
          /*c_mod=*/(short)0, acc[mt], /*reuse_a=*/false, /*reuse_b=*/false);
    }
  }

  // ---- Phase 4: scatter per 16x16 f32 C/D layout (VGPR r, lane-half -> M)
  const int M0   = (lane < 16) ? 0 : 8;
  const int pImg = pbase + waveId * 16 + nIdx;
  float* ob = out + (size_t)b * O_ * HW_ + pImg;
#pragma unroll
  for (int mt = 0; mt < 4; ++mt) {
#pragma unroll
    for (int r = 0; r < 8; ++r) {
      ob[(size_t)(mt * 16 + M0 + r) * HW_] = acc[mt][r];
    }
  }
}

// ---------------------------------------------------------------------------
extern "C" void kernel_launch(void* const* d_in, const int* in_sizes, int n_in,
                              void* d_out, int out_size, void* d_ws, size_t ws_size,
                              hipStream_t stream) {
  const float* x    = (const float*)d_in[0];   // [8, 64, 96, 96] f32
  const float* w    = (const float*)d_in[1];   // [64, 576] f32
  const float* mean = (const float*)d_in[2];   // [1] f32
  const float* stdv = (const float*)d_in[3];   // [1] f32
  float* out        = (float*)d_out;           // [8, 64, 96, 96] f32
  _Float16* wsA     = (_Float16*)d_ws;         // 73,728 B of scratch

  // 1) pre-swizzle weight into WMMA A-fragment layout (f16)
  const int prepN = KSTEPS * 4 * 32 * 16;      // 36,864 halves
  prep_weight_frag<<<(prepN + 255) / 256, 256, 0, stream>>>(w, wsA);

  // 2) fused selection + gather-GEMM
  const int nblocks = (B_ * HW_) / PIXWG;      // 576 workgroups
  pconv_knn_wmma<<<nblocks, THREADS, 0, stream>>>(x, wsA, mean, stdv, out);
}
